// CellWrapper_86741159510122
// MI455X (gfx1250) — matrix operand order repaired and checked by reading, not compiled, not run on hardware
//
#include <hip/hip_runtime.h>
#include <math.h>

namespace {

constexpr int kT    = 4096;  // time steps
constexpr int kB    = 64;    // batch
constexpr int kD    = 128;   // D_in == D_h
constexpr int kRows = 16;    // batch rows per workgroup

typedef __attribute__((ext_vector_type(16))) __bf16        v16bf;
typedef __attribute__((ext_vector_type(2)))  __bf16        v2bf;
typedef __attribute__((ext_vector_type(8)))  float          v8f;
typedef __attribute__((ext_vector_type(16))) unsigned short u16x16;
typedef __attribute__((ext_vector_type(8)))  unsigned short u16x8;
typedef __attribute__((ext_vector_type(4)))  float          f32x4;
typedef __attribute__((ext_vector_type(4)))  unsigned int   u32x4;

// Let the backend select a native cvt (v_cvt_bf16_f32 / v_cvt_pk_bf16_f32)
// for the fptrunc if gfx1250 has it; otherwise clang inlines the RNE sequence.
__device__ __forceinline__ unsigned short f32_to_bf16(float f) {
  return __builtin_bit_cast(unsigned short, (__bf16)f);
}

__device__ __forceinline__ unsigned pk_bf16(float lo, float hi) {
#if __has_builtin(__builtin_amdgcn_cvt_pk_bf16_f32)
  return __builtin_bit_cast(unsigned, __builtin_amdgcn_cvt_pk_bf16_f32(lo, hi));
#else
  v2bf p;
  p[0] = (__bf16)lo;
  p[1] = (__bf16)hi;
  return __builtin_bit_cast(unsigned, p);
#endif
}

__device__ __forceinline__ float fast_tanh(float v) {
#if __has_builtin(__builtin_amdgcn_tanhf)
  return __builtin_amdgcn_tanhf(v);         // v_tanh_f32 (gfx1250)
#else
  return ::tanhf(v);
#endif
}

// Row-major bf16 16x128 tile in LDS -> WMMA A fragment (one 16x32 K-chunk).
// ISA A layout (16-bit, 16x32): lanes 0-15 row M=lane, elements 0..7 = K
// kbase..kbase+7, elements 8..15 = +16, kbase = kc*32 + 8*(lane>=16).
__device__ __forceinline__ v16bf load_a_frag(const unsigned short* p) {
  u16x8 lo = *reinterpret_cast<const u16x8*>(p);        // ds_load_b128
  u16x8 hh = *reinterpret_cast<const u16x8*>(p + 16);   // ds_load_b128
  u16x16 a = __builtin_shufflevector(lo, hh, 0,1,2,3,4,5,6,7,
                                              8,9,10,11,12,13,14,15);
  return __builtin_bit_cast(v16bf, a);
}

} // namespace

__global__ __launch_bounds__(256, 1)
void rnn_scan_wmma(const float* __restrict__ x,
                   const float* __restrict__ Wx,
                   const float* __restrict__ Wh,
                   const float* __restrict__ bias_v,
                   float* __restrict__ out)
{
  __shared__ alignas(16) unsigned short xbuf[2][kRows * kD];
  __shared__ alignas(16) unsigned short hbuf[2][kRows * kD];

  const int tid  = threadIdx.x;
  const int lane = tid & 31;
  const int wv   = tid >> 5;        // 0..7 : output column tile
  const int nloc = lane & 15;       // N within tile
  const int hi   = lane >> 4;       // half-wave select
  const int b0   = blockIdx.x * kRows;

  // ---- one-time: B-operand fragments of Wx, Wh (columns wv*16 .. +15) ----
  // B layout matching A: lane -> N=lane&15, element i -> K = 8*hi + i (i<8), 16 + 8*hi + (i-8) (i>=8).
  const int ncol = wv * 16 + nloc;
  v16bf Bx[4], Bh[4];
#pragma unroll
  for (int kc = 0; kc < 4; ++kc) {
    u16x16 fx, fh;
#pragma unroll
    for (int i = 0; i < 16; ++i) {
      const int k = kc * 32 + hi * 8 + i + (i >= 8 ? 8 : 0);
      fx[i] = f32_to_bf16(Wx[k * kD + ncol]);
      fh[i] = f32_to_bf16(Wh[k * kD + ncol]);
    }
    Bx[kc] = __builtin_bit_cast(v16bf, fx);
    Bh[kc] = __builtin_bit_cast(v16bf, fh);
  }
  const float bias = bias_v[ncol];

  // staging coords: each thread stages 8 consecutive floats of the 16x128 tile
  const int sm = (tid * 8) >> 7;        // row 0..15
  const int sd = (tid * 8) & (kD - 1);  // col, multiple of 8 (16B aligned)
  const float* xs_base = x + (size_t)(b0 + sm) * kT * kD + sd;  // +t*kD per step
  unsigned short* const xslot = &xbuf[0][sm * kD + sd];         // +buffer stride

  // per-lane base into a row-major bf16 tile for A-fragment loads
  const int aoff = nloc * kD + hi * 8;

  // ---- prologue: h0 = 0; stage x[0]; preload x[1] into registers ----
  for (int i = tid; i < kRows * kD; i += 256) hbuf[0][i] = 0;
  {
    const f32x4* s4 = reinterpret_cast<const f32x4*>(xs_base);
    f32x4 a = s4[0], b = s4[1];
    u32x4 v;
    v[0] = pk_bf16(a[0], a[1]); v[1] = pk_bf16(a[2], a[3]);
    v[2] = pk_bf16(b[0], b[1]); v[3] = pk_bf16(b[2], b[3]);
    *reinterpret_cast<u32x4*>(xslot) = v;
  }
  f32x4 r0 = {}, r1 = {};
  {
    const f32x4* s4 = reinterpret_cast<const f32x4*>(xs_base + kD);
    r0 = s4[0]; r1 = s4[1];
  }
  __syncthreads();

  const size_t out_lane_base = (size_t)b0 * kT * kD + (size_t)ncol;

  int cur = 0;
  for (int t = 0; t < kT; ++t) {
    const int nxt = cur ^ 1;

    // --- critical path first: h_{t-1} fragment loads go out immediately ---
    const unsigned short* hb = &hbuf[cur][aoff];
    v16bf ah0 = load_a_frag(hb);
    v16bf ah1 = load_a_frag(hb + 32);
    v16bf ah2 = load_a_frag(hb + 64);
    v16bf ah3 = load_a_frag(hb + 96);

    // --- off-path: stage x[t+1] from regs (loaded last iter), start x[t+2] ---
    if (t + 1 < kT) {
      u32x4 v;
      v[0] = pk_bf16(r0[0], r0[1]); v[1] = pk_bf16(r0[2], r0[3]);
      v[2] = pk_bf16(r1[0], r1[1]); v[3] = pk_bf16(r1[2], r1[3]);
      *reinterpret_cast<u32x4*>(xslot + (nxt ? kRows * kD : 0)) = v;
    }
    if (t + 2 < kT) {
      const f32x4* s4 =
          reinterpret_cast<const f32x4*>(xs_base + (size_t)(t + 2) * kD);
      r0 = s4[0]; r1 = s4[1];
    }
    if (t + 8 < kT)
      __builtin_prefetch(xs_base + (size_t)(t + 8) * kD, 0, 3);

    // --- x_t @ Wx : two independent 2-deep chains (off the recurrence) ---
    const unsigned short* xb = &xbuf[cur][aoff];
    v8f acc0 = {};
    v8f acc1 = {};
    acc0 = __builtin_amdgcn_wmma_f32_16x16x32_bf16(
        false, load_a_frag(xb),      false, Bx[0], (short)0, acc0, false, false);
    acc1 = __builtin_amdgcn_wmma_f32_16x16x32_bf16(
        false, load_a_frag(xb + 64), false, Bx[2], (short)0, acc1, false, false);
    acc0 = __builtin_amdgcn_wmma_f32_16x16x32_bf16(
        false, load_a_frag(xb + 32), false, Bx[1], (short)0, acc0, false, false);
    acc1 = __builtin_amdgcn_wmma_f32_16x16x32_bf16(
        false, load_a_frag(xb + 96), false, Bx[3], (short)0, acc1, false, false);

    // --- + h_{t-1} @ Wh : serial part is only 2 chained WMMAs per chain ---
    acc0 = __builtin_amdgcn_wmma_f32_16x16x32_bf16(
        false, ah0, false, Bh[0], (short)0, acc0, false, false);
    acc1 = __builtin_amdgcn_wmma_f32_16x16x32_bf16(
        false, ah2, false, Bh[2], (short)0, acc1, false, false);
    acc0 = __builtin_amdgcn_wmma_f32_16x16x32_bf16(
        false, ah1, false, Bh[1], (short)0, acc0, false, false);
    acc1 = __builtin_amdgcn_wmma_f32_16x16x32_bf16(
        false, ah3, false, Bh[3], (short)0, acc1, false, false);

    // --- h_t = tanh(z); write hbuf[nxt] (bf16, serial path) and out (f32) ---
    float hv[8];
#pragma unroll
    for (int r = 0; r < 8; ++r)
      hv[r] = fast_tanh(acc0[r] + acc1[r] + bias);

    unsigned short* hn = &hbuf[nxt][(size_t)hi * 8 * kD + ncol];
#pragma unroll
    for (int r = 0; r < 8; r += 2) {
      const unsigned p = pk_bf16(hv[r], hv[r + 1]);
      hn[(size_t)r * kD]       = (unsigned short)p;         // ds_store_b16
      hn[(size_t)(r + 1) * kD] = (unsigned short)(p >> 16); // d16_hi store
    }
#pragma unroll
    for (int r = 0; r < 8; ++r) {
      const int m = r + hi * 8;  // C/D layout: VGPR r -> M = r + 8*(lane>=16)
      out[out_lane_base + ((size_t)m * kT + (size_t)t) * kD] = hv[r];
    }

    __syncthreads();
    cur = nxt;
  }
}

extern "C" void kernel_launch(void* const* d_in, const int* in_sizes, int n_in,
                              void* d_out, int out_size, void* d_ws, size_t ws_size,
                              hipStream_t stream) {
  (void)in_sizes; (void)n_in; (void)out_size; (void)d_ws; (void)ws_size;
  const float* x  = (const float*)d_in[0];
  const float* Wx = (const float*)d_in[1];
  const float* Wh = (const float*)d_in[2];
  const float* b  = (const float*)d_in[3];
  float* out = (float*)d_out;

  dim3 grid(kB / kRows);  // 4 workgroups, one per 16 batch rows
  dim3 block(256);        // 8 wave32 waves -> 8 WMMA column tiles
  hipLaunchKernelGGL(rnn_scan_wmma, grid, block, 0, stream, x, Wx, Wh, b, out);
}
